// BigBirdAttention_71914932404458
// MI455X (gfx1250) — matrix operand order repaired and checked by reading, not compiled, not run on hardware
//
#include <hip/hip_runtime.h>

// ---------------------------------------------------------------------------
// BigBird attention for MI455X (gfx1250), wave32 + WMMA f32<-f16 16x16x32.
//
// Roofline: useful traffic = q/k/v (50 MB) + touched mask slices (84 MB of the
// 536 MB tensor) + out (17 MB) ~= 150 MB @ 23.3 TB/s ~= 6.5 us.  All matmuls
// (~4.8 GF) ride v_wmma_f32_16x16x32_f16.  bb_diag has 512 WGs; the global and
// random passes are split-K over 4 WGs per (b,h) (128 WGs each) with a
// log-sum-exp combine, so no kernel leaves most WGPs idle.
// ---------------------------------------------------------------------------

typedef __attribute__((ext_vector_type(16))) _Float16 v16h;
typedef __attribute__((ext_vector_type(8)))  float    v8f;

constexpr int B_ = 2, S_ = 2048, H_ = 16, D_ = 64;
constexpr int BS_ = 128, NB_ = 16, R_ = 64;
constexpr float SCALE_ = 0.125f;       // 1/sqrt(64)

constexpr int KT_STR  = 72;            // f16 row stride for K/V LDS tiles
constexpr int WT_STR  = 136;           // f16 row stride for probability tiles
constexpr int REGION0 = 8 * 16 * WT_STR;   // 17408 halfs; also covers kT (128*72)
constexpr int REGION1 = 128 * KT_STR;      // vT

// Split-K workspace layout (floats):
constexpr int SPLIT_ = 4;
constexpr int KB_SPLIT_ = NB_ / SPLIT_;            // 4 key blocks per split
constexpr int GP_STRIDE = BS_ * D_ + 2 * BS_;      // N(128x64) + m(128) + z(128)
constexpr long long GP_TOTAL = (long long)B_ * H_ * SPLIT_ * GP_STRIDE;
constexpr int RP_STRIDE = 2 * R_;                  // m(64) + z(64)
constexpr long long WS_FLOATS =
    GP_TOTAL + (long long)B_ * H_ * SPLIT_ * RP_STRIDE;

__device__ __forceinline__ size_t qix(int b, int s, int h, int d) {
  return (((size_t)b * S_ + s) * H_ + h) * D_ + d;
}
__device__ __forceinline__ size_t mix(int b, int h, int qi, int ki) {
  return (((size_t)b * H_ + h) * S_ + qi) * S_ + ki;
}

// A-fragment (16x32 f16, MxK): lane 0-15 -> row M=lane; element e ->
// K = 16*(lane>=16) + e.
__device__ __forceinline__ int afk(int lane, int e) {
  return (lane & 16) | e;
}
// B-fragment (32x16 f16, KxN): lane holds column N=lane&15; lanes 0-15 carry
// K=0..15 sequentially (element e), lanes 16-31 carry K=16..31.

__device__ __forceinline__ float gmax16(float x) {
#pragma unroll
  for (int o = 1; o < 16; o <<= 1) x = fmaxf(x, __shfl_xor(x, o, 32));
  return x;
}
__device__ __forceinline__ float gsum16(float x) {
#pragma unroll
  for (int o = 1; o < 16; o <<= 1) x += __shfl_xor(x, o, 32);
  return x;
}

__device__ __forceinline__ v8f wmma_f16(v16h a, v16h b, v8f c) {
  return __builtin_amdgcn_wmma_f32_16x16x32_f16(false, a, false, b, (short)0, c,
                                                false, false);
}

__device__ __forceinline__ void load_q_frags(const float* __restrict__ q, int b,
                                             int h, int row, int lane, v16h& aq0,
                                             v16h& aq1) {
#pragma unroll
  for (int e = 0; e < 16; ++e) {
    int kk = afk(lane, e);
    aq0[e] = (_Float16)q[qix(b, row, h, kk)];
    aq1[e] = (_Float16)q[qix(b, row, h, kk + 32)];
  }
}

// ---------------------------------------------------------------------------
// Shared flash-attention inner loop over key blocks [kb0, kb1).
// ridx[8]: global mask-row ids in C layout.  DO_PV keeps a rescaled PV
// accumulator (oa) via an LDS-staged probability tile; otherwise only the
// online (rowmax, rowsum) statistics are maintained.
// ---------------------------------------------------------------------------
template <int NT, bool DO_PV>
__device__ __forceinline__ void flash_keys(
    const float* __restrict__ k, const float* __restrict__ v,
    const float* __restrict__ mask, int b, int h, int kb0, int kb1, v16h aq0,
    v16h aq1, const int ridx[8], float rowmax[8], float rowsum[8], v8f* oa,
    _Float16* kT, _Float16* wT, _Float16* vT) {
  const int tid = threadIdx.x;
  const int lane = tid & 31;
  const int w = tid >> 5;
  const int m = lane & 15;
  const int hi = (lane & 16) ? 1 : 0;

  for (int kb = kb0; kb < kb1; ++kb) {
    __syncthreads();  // previous iteration's LDS reads complete
    const int base = kb * BS_;
    for (int idx = tid; idx < BS_ * D_; idx += NT) {
      int r = idx >> 6, d = idx & 63;
      kT[r * KT_STR + d] = (_Float16)k[qix(b, base + r, h, d)];
      if constexpr (DO_PV)
        vT[r * KT_STR + d] = (_Float16)v[qix(b, base + r, h, d)];
    }
    __syncthreads();

    float sc[8][8];
#pragma unroll
    for (int j = 0; j < 8; ++j) {
      v16h bk0, bk1;
#pragma unroll
      for (int e = 0; e < 16; ++e) {
        int dd = (lane & 16) + e;
        bk0[e] = kT[(16 * j + m) * KT_STR + dd];
        bk1[e] = kT[(16 * j + m) * KT_STR + dd + 32];
      }
      v8f acc = {};
      acc = wmma_f16(aq0, bk0, acc);
      acc = wmma_f16(aq1, bk1, acc);
#pragma unroll
      for (int r = 0; r < 8; ++r)
        sc[j][r] = acc[r] * SCALE_ + mask[mix(b, h, ridx[r], base + 16 * j + m)];
    }
    if constexpr (DO_PV) __syncthreads();  // kT reads done before wT overwrite

#pragma unroll
    for (int r = 0; r < 8; ++r) {
      float tm = -3.0e38f;
#pragma unroll
      for (int j = 0; j < 8; ++j) tm = fmaxf(tm, sc[j][r]);
      tm = gmax16(tm);
      float nm = fmaxf(rowmax[r], tm);
      float corr = __expf(rowmax[r] - nm);  // 0 on first block
      rowmax[r] = nm;
      float ps = 0.f;
#pragma unroll
      for (int j = 0; j < 8; ++j) {
        float p = __expf(sc[j][r] - nm);
        sc[j][r] = p;
        ps += p;
      }
      rowsum[r] = rowsum[r] * corr + gsum16(ps);
      if constexpr (DO_PV) {
#pragma unroll
        for (int c = 0; c < 4; ++c) oa[c][r] *= corr;
        int rl = w * 16 + r + hi * 8;
#pragma unroll
        for (int j = 0; j < 8; ++j)
          wT[rl * WT_STR + 16 * j + m] = (_Float16)sc[j][r];
      }
    }
    if constexpr (DO_PV) {
      __syncthreads();
#pragma unroll
      for (int t = 0; t < 4; ++t) {
        v16h aw;
#pragma unroll
        for (int e = 0; e < 16; ++e)
          aw[e] = wT[(w * 16 + m) * WT_STR + 32 * t + afk(lane, e)];
#pragma unroll
        for (int c = 0; c < 4; ++c) {
          v16h bv;
#pragma unroll
          for (int e = 0; e < 16; ++e)
            bv[e] = vT[(32 * t + (lane & 16) + e) * KT_STR + 16 * c + m];
          oa[c] = wmma_f16(aw, bv, oa[c]);
        }
      }
    }
  }
}

// ---------------------------------------------------------------------------
// Kernel 1: block-diagonal attention.  One WG per (b, block, h); writes out.
// ---------------------------------------------------------------------------
__global__ __launch_bounds__(256) void bb_diag(const float* __restrict__ q,
                                               const float* __restrict__ k,
                                               const float* __restrict__ v,
                                               const float* __restrict__ mask,
                                               float* __restrict__ out) {
  __shared__ _Float16 sm[REGION0 + REGION1];
  _Float16* kT = sm;            // phase 1
  _Float16* wT = sm;            // phase 2 (reuses kT space)
  _Float16* vT = sm + REGION0;

  const int bid = blockIdx.x;
  const int b = bid / (NB_ * H_);
  const int n = (bid / H_) % NB_;
  const int h = bid % H_;
  const int base = n * BS_;
  const int tid = threadIdx.x;
  const int lane = tid & 31;
  const int w = tid >> 5;
  const int m = lane & 15;
  const int hi = (lane & 16) ? 1 : 0;

  for (int idx = tid; idx < BS_ * D_; idx += 256) {
    int r = idx >> 6, d = idx & 63;
    kT[r * KT_STR + d] = (_Float16)k[qix(b, base + r, h, d)];
    vT[r * KT_STR + d] = (_Float16)v[qix(b, base + r, h, d)];
  }
  __syncthreads();

  v16h aq0, aq1;
  load_q_frags(q, b, h, base + w * 16 + m, lane, aq0, aq1);

  float sc[8][8];
#pragma unroll
  for (int j = 0; j < 8; ++j) {
    v16h bk0, bk1;
#pragma unroll
    for (int e = 0; e < 16; ++e) {
      int dd = (lane & 16) + e;
      bk0[e] = kT[(16 * j + m) * KT_STR + dd];
      bk1[e] = kT[(16 * j + m) * KT_STR + dd + 32];
    }
    v8f acc = {};
    acc = wmma_f16(aq0, bk0, acc);
    acc = wmma_f16(aq1, bk1, acc);
#pragma unroll
    for (int r = 0; r < 8; ++r) {
      int gq = base + w * 16 + r + hi * 8;
      int gk = base + 16 * j + m;
      sc[j][r] = acc[r] * SCALE_ + mask[mix(b, h, gq, gk)];
    }
  }
  __syncthreads();  // all waves done reading kT before wT overwrites it

#pragma unroll
  for (int r = 0; r < 8; ++r) {
    float rm = -3.0e38f;
#pragma unroll
    for (int j = 0; j < 8; ++j) rm = fmaxf(rm, sc[j][r]);
    rm = gmax16(rm);
    float s = 0.f;
#pragma unroll
    for (int j = 0; j < 8; ++j) {
      float p = __expf(sc[j][r] - rm);
      sc[j][r] = p;
      s += p;
    }
    s = gsum16(s);
    float inv = 1.0f / s;
    int rl = w * 16 + r + hi * 8;
#pragma unroll
    for (int j = 0; j < 8; ++j)
      wT[rl * WT_STR + 16 * j + m] = (_Float16)(sc[j][r] * inv);
  }
  __syncthreads();

  v8f oa[4] = {};
#pragma unroll
  for (int t = 0; t < 4; ++t) {
    v16h aw;
#pragma unroll
    for (int e = 0; e < 16; ++e)
      aw[e] = wT[(w * 16 + m) * WT_STR + 32 * t + afk(lane, e)];
#pragma unroll
    for (int c = 0; c < 4; ++c) {
      v16h bv;
#pragma unroll
      for (int e = 0; e < 16; ++e)
        bv[e] = vT[(32 * t + (lane & 16) + e) * KT_STR + 16 * c + m];
      oa[c] = wmma_f16(aw, bv, oa[c]);
    }
  }
#pragma unroll
  for (int c = 0; c < 4; ++c)
#pragma unroll
    for (int r = 0; r < 8; ++r)
      out[qix(b, base + w * 16 + r + hi * 8, h, 16 * c + m)] = oa[c][r];
}

// ---------------------------------------------------------------------------
// Global attention (first 128 queries vs all keys).
// Split-K version: one WG per (b,h,split); partials into ws.
// ---------------------------------------------------------------------------
__global__ __launch_bounds__(256) void bb_global_part(
    const float* __restrict__ q, const float* __restrict__ k,
    const float* __restrict__ v, const float* __restrict__ mask,
    float* __restrict__ ws) {
  __shared__ _Float16 sm[REGION0 + REGION1];
  const int sp = blockIdx.x % SPLIT_;
  const int bh = blockIdx.x / SPLIT_;
  const int b = bh / H_, h = bh % H_;
  const int tid = threadIdx.x, lane = tid & 31, w = tid >> 5;
  const int m = lane & 15;
  const int hi = (lane & 16) ? 1 : 0;

  int ridx[8];
#pragma unroll
  for (int r = 0; r < 8; ++r) ridx[r] = w * 16 + r + hi * 8;

  v16h aq0, aq1;
  load_q_frags(q, b, h, w * 16 + m, lane, aq0, aq1);

  float rowmax[8], rowsum[8];
#pragma unroll
  for (int r = 0; r < 8; ++r) { rowmax[r] = -3.0e38f; rowsum[r] = 0.f; }
  v8f oa[4] = {};

  flash_keys<256, true>(k, v, mask, b, h, sp * KB_SPLIT_, (sp + 1) * KB_SPLIT_,
                        aq0, aq1, ridx, rowmax, rowsum, oa, sm, sm, sm + REGION0);

  float* p = ws + ((size_t)bh * SPLIT_ + sp) * GP_STRIDE;
#pragma unroll
  for (int c = 0; c < 4; ++c)
#pragma unroll
    for (int r = 0; r < 8; ++r)
      p[(size_t)ridx[r] * 64 + 16 * c + m] = oa[c][r];  // unnormalized N
  if (m == 0) {
#pragma unroll
    for (int r = 0; r < 8; ++r) {
      p[BS_ * D_ + ridx[r]] = rowmax[r];
      p[BS_ * D_ + BS_ + ridx[r]] = rowsum[r];
    }
  }
}

__global__ __launch_bounds__(256) void bb_global_comb(const float* __restrict__ ws,
                                                      float* __restrict__ out) {
  const int b = blockIdx.x / H_, h = blockIdx.x % H_;
  const float* p0 = ws + (size_t)blockIdx.x * SPLIT_ * GP_STRIDE;
  for (int e = threadIdx.x; e < BS_ * D_; e += 256) {
    int g = e >> 6, d = e & 63;
    float M = -3.0e38f;
#pragma unroll
    for (int sp = 0; sp < SPLIT_; ++sp)
      M = fmaxf(M, p0[sp * GP_STRIDE + BS_ * D_ + g]);
    float Z = 0.f, N = 0.f;
#pragma unroll
    for (int sp = 0; sp < SPLIT_; ++sp) {
      float wgt = __expf(p0[sp * GP_STRIDE + BS_ * D_ + g] - M);
      Z += p0[sp * GP_STRIDE + BS_ * D_ + BS_ + g] * wgt;
      N += p0[sp * GP_STRIDE + e] * wgt;
    }
    out[qix(b, g, h, d)] += N / Z;  // disjoint per (b,h); after bb_diag
  }
}

// Monolithic fallback (used only if ws is too small).
__global__ __launch_bounds__(256) void bb_global(const float* __restrict__ q,
                                                 const float* __restrict__ k,
                                                 const float* __restrict__ v,
                                                 const float* __restrict__ mask,
                                                 float* __restrict__ out) {
  __shared__ _Float16 sm[REGION0 + REGION1];
  const int b = blockIdx.x / H_, h = blockIdx.x % H_;
  const int tid = threadIdx.x, lane = tid & 31, w = tid >> 5;
  const int m = lane & 15;
  const int hi = (lane & 16) ? 1 : 0;

  int ridx[8];
#pragma unroll
  for (int r = 0; r < 8; ++r) ridx[r] = w * 16 + r + hi * 8;

  v16h aq0, aq1;
  load_q_frags(q, b, h, w * 16 + m, lane, aq0, aq1);

  float rowmax[8], rowsum[8];
#pragma unroll
  for (int r = 0; r < 8; ++r) { rowmax[r] = -3.0e38f; rowsum[r] = 0.f; }
  v8f oa[4] = {};

  flash_keys<256, true>(k, v, mask, b, h, 0, NB_, aq0, aq1, ridx, rowmax,
                        rowsum, oa, sm, sm, sm + REGION0);

#pragma unroll
  for (int r = 0; r < 8; ++r) {
    float inv = 1.0f / rowsum[r];
#pragma unroll
    for (int c = 0; c < 4; ++c)
      out[qix(b, ridx[r], h, 16 * c + m)] += oa[c][r] * inv;
  }
}

// ---------------------------------------------------------------------------
// Random rows: only the softmax rowsum is needed (out += v * rowsum,
// rowsum = Z*(1/Z) ~= 1), so track (rowmax, Z) only -- no PV matmul.
// Split-K version + combine; monolithic fallback below.
// ---------------------------------------------------------------------------
__global__ __launch_bounds__(128) void bb_rand_part(const float* __restrict__ q,
                                                    const float* __restrict__ k,
                                                    const float* __restrict__ mask,
                                                    const int* __restrict__ rind,
                                                    float* __restrict__ ws) {
  __shared__ _Float16 kTs[128 * KT_STR];
  const int sp = blockIdx.x % SPLIT_;
  const int bh = blockIdx.x / SPLIT_;
  const int b = bh / H_, h = bh % H_;
  const int tid = threadIdx.x, lane = tid & 31, w = tid >> 5;  // w in 0..3
  const int m = lane & 15;
  const int hi = (lane & 16) ? 1 : 0;

  const int qa = rind[w * 16 + m];
  int ridx[8];
#pragma unroll
  for (int r = 0; r < 8; ++r) ridx[r] = rind[w * 16 + r + hi * 8];

  v16h aq0, aq1;
  load_q_frags(q, b, h, qa, lane, aq0, aq1);

  float rowmax[8], rowz[8];
#pragma unroll
  for (int r = 0; r < 8; ++r) { rowmax[r] = -3.0e38f; rowz[r] = 0.f; }
  v8f dummy[1];

  flash_keys<128, false>(k, nullptr, mask, b, h, sp * KB_SPLIT_,
                         (sp + 1) * KB_SPLIT_, aq0, aq1, ridx, rowmax, rowz,
                         dummy, kTs, kTs, kTs);

  float* p = ws + GP_TOTAL + ((size_t)bh * SPLIT_ + sp) * RP_STRIDE;
  if (m == 0) {
#pragma unroll
    for (int r = 0; r < 8; ++r) {
      int row = w * 16 + r + hi * 8;  // 0..63
      p[row] = rowmax[r];
      p[R_ + row] = rowz[r];
    }
  }
}

__global__ __launch_bounds__(256) void bb_rand_comb(const float* __restrict__ ws,
                                                    const float* __restrict__ v,
                                                    const int* __restrict__ rind,
                                                    float* __restrict__ out) {
  const int b = blockIdx.x / H_, h = blockIdx.x % H_;
  const float* p0 = ws + GP_TOTAL + (size_t)blockIdx.x * SPLIT_ * RP_STRIDE;
  for (int e = threadIdx.x; e < R_ * D_; e += 256) {
    int row = e >> 6, d = e & 63;
    float M = -3.0e38f;
#pragma unroll
    for (int sp = 0; sp < SPLIT_; ++sp)
      M = fmaxf(M, p0[sp * RP_STRIDE + row]);
    float Z = 0.f;
#pragma unroll
    for (int sp = 0; sp < SPLIT_; ++sp)
      Z += p0[sp * RP_STRIDE + R_ + row] * __expf(p0[sp * RP_STRIDE + row] - M);
    float rs = Z * (1.0f / Z);  // softmax row sum
    int qi = rind[row];
    atomicAdd(&out[qix(b, qi, h, d)], v[qix(b, qi, h, d)] * rs);  // dups possible
  }
}

// Monolithic fallback.
__global__ __launch_bounds__(128) void bb_rand(const float* __restrict__ q,
                                               const float* __restrict__ k,
                                               const float* __restrict__ v,
                                               const float* __restrict__ mask,
                                               const int* __restrict__ rind,
                                               float* __restrict__ out) {
  __shared__ _Float16 kTs[128 * KT_STR];
  const int b = blockIdx.x / H_, h = blockIdx.x % H_;
  const int tid = threadIdx.x, lane = tid & 31, w = tid >> 5;
  const int m = lane & 15;
  const int hi = (lane & 16) ? 1 : 0;

  const int qa = rind[w * 16 + m];
  int ridx[8];
#pragma unroll
  for (int r = 0; r < 8; ++r) ridx[r] = rind[w * 16 + r + hi * 8];

  v16h aq0, aq1;
  load_q_frags(q, b, h, qa, lane, aq0, aq1);

  float rowmax[8], rowz[8];
#pragma unroll
  for (int r = 0; r < 8; ++r) { rowmax[r] = -3.0e38f; rowz[r] = 0.f; }
  v8f dummy[1];

  flash_keys<128, false>(k, nullptr, mask, b, h, 0, NB_, aq0, aq1, ridx, rowmax,
                         rowz, dummy, kTs, kTs, kTs);

#pragma unroll
  for (int r = 0; r < 8; ++r) {
    float z = rowz[r];
    float rs = z * (1.0f / z);
    int qi = ridx[r];
#pragma unroll
    for (int t = 0; t < 4; ++t) {
      int d = m * 4 + t;
      atomicAdd(&out[qix(b, qi, h, d)], v[qix(b, qi, h, d)] * rs);
    }
  }
}

// ---------------------------------------------------------------------------
extern "C" void kernel_launch(void* const* d_in, const int* in_sizes, int n_in,
                              void* d_out, int out_size, void* d_ws,
                              size_t ws_size, hipStream_t stream) {
  (void)in_sizes; (void)n_in; (void)out_size;
  const float* q = (const float*)d_in[0];
  const float* k = (const float*)d_in[1];
  const float* v = (const float*)d_in[2];
  const float* mask = (const float*)d_in[3];
  const int* rind = (const int*)d_in[4];
  float* out = (float*)d_out;

  bb_diag<<<B_ * NB_ * H_, 256, 0, stream>>>(q, k, v, mask, out);  // writes out

  const bool split_ok =
      d_ws != nullptr && ws_size >= (size_t)WS_FLOATS * sizeof(float);
  if (split_ok) {
    float* ws = (float*)d_ws;
    bb_global_part<<<B_ * H_ * SPLIT_, 256, 0, stream>>>(q, k, v, mask, ws);
    bb_rand_part<<<B_ * H_ * SPLIT_, 128, 0, stream>>>(q, k, mask, rind, ws);
    bb_global_comb<<<B_ * H_, 256, 0, stream>>>(ws, out);
    bb_rand_comb<<<B_ * H_, 256, 0, stream>>>(ws, v, rind, out);
  } else {
    bb_global<<<B_ * H_, 256, 0, stream>>>(q, k, v, mask, out);
    bb_rand<<<B_ * H_, 128, 0, stream>>>(q, k, v, mask, rind, out);
  }
}
